// MutationAwareGNN_11630771438397
// MI455X (gfx1250) — matrix operand order repaired and checked by reading, not compiled, not run on hardware
//
#include <hip/hip_runtime.h>
#include <hip/hip_bf16.h>

// ---------------------------------------------------------------------------
// GCN (3x GCNConv + mean-pool + MLP head) for MI455X / gfx1250, wave32.
// Dense transforms: V_WMMA_F32_16X16X32_F16, f16 activations (pre-converted),
// weights staged in LDS as packed f16 k-pairs -> ds_load_b128 B fragments,
// global_load_b128 A fragments, no guards (M%16==0, K%32==0 by padding).
// Sparse aggregation: L2-resident float4 gathers + global_atomic_add_f32.
// ---------------------------------------------------------------------------

typedef __attribute__((ext_vector_type(16))) _Float16 v16h;
typedef __attribute__((ext_vector_type(8)))  float    v8f;

#define GCN_N      100000
#define GCN_E      3200000
#define GCN_G      256
#define GCN_INDIM  25
#define GCN_HID    64
#define GCN_OUT3   32

// ----------------------------- utility kernels -----------------------------

__global__ void k_fill(float* __restrict__ p, float v, int n) {
    int i = blockIdx.x * blockDim.x + threadIdx.x;
    if (i < n) p[i] = v;
}

// deg[dst] += 1 per edge (buffer pre-filled with 1.0 for the self-loop)
__global__ void k_degree(const int* __restrict__ dst, float* __restrict__ deg, int E) {
    int e = blockIdx.x * blockDim.x + threadIdx.x;
    if (e < E) atomicAdd(&deg[dst[e]], 1.0f);
}

// deg -> d^{-1/2} in place
__global__ void k_dinv(float* __restrict__ d, int n) {
    int i = blockIdx.x * blockDim.x + threadIdx.x;
    if (i < n) d[i] = rsqrtf(fmaxf(d[i], 1.0f));
}

// pad/convert x [N,25] f32 -> xh [N,32] f16 (zeros in k>=25)
__global__ void k_xpad(const float* __restrict__ x, _Float16* __restrict__ xh, int n) {
    int t = blockIdx.x * blockDim.x + threadIdx.x;
    if (t < (n << 5)) {
        const int i = t >> 5;
        const int k = t & 31;
        xh[t] = (k < GCN_INDIM) ? (_Float16)x[i * GCN_INDIM + k] : (_Float16)0.0f;
    }
}

// ----------------------------- WMMA GEMM -----------------------------------
// Y[M,NC](f32) = Xh[M,K](f16) @ W[K,NC](f32)  (+bias, optional relu).
// One wave computes a 16 x NC row-stripe (NT = NC/16 accumulators).
// Requirements: M % 16 == 0, K % 32 == 0 (Kact <= K rows are real, rest zero).
// Fragment layouts per CDNA5 ISA 7.12.2 (wave32):
//   A 16x32 f16 : lane m=lane&15, hi=lane>>4; frag halves [0..7]  = K kk+hi*8..+7,
//                 frag halves [8..15] = K kk+16+hi*8..+7  -> two b128 loads.
//   B 32x16 f16 : lane n=lane&15, hi=lane>>4; frag pairs [0..3] = K kk+hi*8.., [4..7] = K kk+16+hi*8..
//                 -> packed u32 per pair, col-major in LDS -> two ds b128 loads.
//   C/D 16x16 f32: VGPR r: M = r + 8*hi, N = lane&15.
template<int K, int NC>
__global__ __launch_bounds__(32)
void k_gemm_wmma(const _Float16* __restrict__ Xh, const float* __restrict__ W, int Kact,
                 float* __restrict__ Y, const float* __restrict__ bias, int relu)
{
    constexpr int KP = K / 2;     // packed f16 k-pairs per column
    constexpr int NT = NC / 16;   // 16-wide column tiles

    __shared__ __align__(16) unsigned Wl[NC * KP];   // max 64*32*4B = 8 KB

    const int lane  = threadIdx.x;
    const int tileM = blockIdx.x * 16;

    // stage W (f32 [K,NC] row-major) into LDS: col-major packed f16 pairs
    for (int idx = lane; idx < NC * KP; idx += 32) {
        const int col = idx / KP;
        const int kp  = idx % KP;
        union { _Float16 h[2]; unsigned u; } pk;
        pk.h[0] = (2 * kp     < Kact) ? (_Float16)W[(2 * kp)     * NC + col] : (_Float16)0.0f;
        pk.h[1] = (2 * kp + 1 < Kact) ? (_Float16)W[(2 * kp + 1) * NC + col] : (_Float16)0.0f;
        Wl[idx] = pk.u;
    }
    __syncthreads();

    const int n  = lane & 15;
    const int hi = lane >> 4;

    v8f acc[NT];
    #pragma unroll
    for (int t = 0; t < NT; ++t) acc[t] = (v8f){};

    const _Float16* Xr = Xh + (long)(tileM + n) * K;   // A row for this lane

    #pragma unroll
    for (int kk = 0; kk < K; kk += 32) {
        union { v16h h; float4 f[2]; } au;
        au.f[0] = *(const float4*)(Xr + kk + hi * 8);        // halves k..k+7
        au.f[1] = *(const float4*)(Xr + kk + 16 + hi * 8);   // halves k+16..k+23
        #pragma unroll
        for (int t = 0; t < NT; ++t) {
            union { v16h h; float4 f[2]; } bu;
            const unsigned* wp = &Wl[(t * 16 + n) * KP + (kk >> 1) + hi * 4];
            bu.f[0] = *(const float4*)(wp);
            bu.f[1] = *(const float4*)(wp + 8);
            acc[t] = __builtin_amdgcn_wmma_f32_16x16x32_f16(false, au.h, false, bu.h,
                                                            (short)0, acc[t], false, false);
        }
    }

    #pragma unroll
    for (int t = 0; t < NT; ++t) {
        const int ocol = t * 16 + n;
        float badd = bias ? bias[ocol] : 0.0f;
        #pragma unroll
        for (int r = 0; r < 8; ++r) {
            const int orow = tileM + r + 8 * hi;
            float v = acc[t][r] + badd;
            if (relu) v = fmaxf(v, 0.0f);
            Y[(long)orow * NC + ocol] = v;
        }
    }
}

// ------------------------- sparse aggregation ------------------------------

// A[i,:] = H[i,:] * dinv[i]^2   (self-loop term; also initializes the buffer)
__global__ void k_selfloop(const float* __restrict__ H, const float* __restrict__ dinv,
                           float* __restrict__ A, int n, int logF) {
    int t = blockIdx.x * blockDim.x + threadIdx.x;
    if (t < (n << logF)) {
        const int i = t >> logF;
        const float s = dinv[i];
        A[t] = H[t] * s * s;
    }
}

// A[dst,:] += H[src,:] * dinv[src]*dinv[dst]
// F/16 threads per edge; each handles 16 features (4 x float4 gather + 16 atomics)
__global__ void k_edge_agg(const int* __restrict__ src, const int* __restrict__ dstv,
                           const float* __restrict__ H, const float* __restrict__ dinv,
                           float* __restrict__ A, int E, int logT, int logF) {
    int t = blockIdx.x * blockDim.x + threadIdx.x;
    if (t >= (E << logT)) return;
    const int e = t >> logT;
    const int c = t & ((1 << logT) - 1);
    const int s = src[e];
    const int d = dstv[e];
    const float norm = dinv[s] * dinv[d];
    const float4* hp = (const float4*)(H + ((long)s << logF)) + (c << 2);
    float*        o  = A + ((long)d << logF) + (c << 4);
    #pragma unroll
    for (int j = 0; j < 4; ++j) {
        const float4 v = hp[j];
        atomicAdd(o + 4 * j + 0, v.x * norm);
        atomicAdd(o + 4 * j + 1, v.y * norm);
        atomicAdd(o + 4 * j + 2, v.z * norm);
        atomicAdd(o + 4 * j + 3, v.w * norm);
    }
}

// A = relu(A + bias[f]) in place (f32) and as f16 copy for the next GEMM's A
__global__ void k_bias_relu_h(float* __restrict__ A, _Float16* __restrict__ Ah,
                              const float* __restrict__ bias, int n, int logF) {
    int t = blockIdx.x * blockDim.x + threadIdx.x;
    if (t < (n << logF)) {
        const int f = t & ((1 << logF) - 1);
        const float v = fmaxf(A[t] + bias[f], 0.0f);
        A[t]  = v;
        Ah[t] = (_Float16)v;
    }
}

// ------------------------------ pooling + head -----------------------------

__global__ void k_pool_accum(const float* __restrict__ H, const int* __restrict__ batch,
                             float* __restrict__ gsum, float* __restrict__ gcnt, int n) {
    int t = blockIdx.x * blockDim.x + threadIdx.x;   // n * 2 threads (16 feats each)
    if (t >= (n << 1)) return;
    const int i = t >> 1;
    const int c = t & 1;
    const int g = batch[i];
    const float4* hp = (const float4*)(H + (long)i * 32) + (c << 2);
    float*        o  = gsum + g * 32 + (c << 4);
    #pragma unroll
    for (int j = 0; j < 4; ++j) {
        const float4 v = hp[j];
        atomicAdd(o + 4 * j + 0, v.x);
        atomicAdd(o + 4 * j + 1, v.y);
        atomicAdd(o + 4 * j + 2, v.z);
        atomicAdd(o + 4 * j + 3, v.w);
    }
    if (c == 0) atomicAdd(&gcnt[g], 1.0f);
}

__global__ void k_pool_div_h(float* __restrict__ gsum, const float* __restrict__ gcnt,
                             _Float16* __restrict__ gh, int nG) {
    int t = blockIdx.x * blockDim.x + threadIdx.x;
    if (t < nG * 32) {
        const int g = t >> 5;
        const float v = gsum[t] / fmaxf(gcnt[g], 1.0f);
        gsum[t] = v;
        gh[t]   = (_Float16)v;
    }
}

__global__ void k_head2(const float* __restrict__ hid, const float* __restrict__ Wh2,
                        const float* __restrict__ bh2, float* __restrict__ y, int nG) {
    int g = blockIdx.x * blockDim.x + threadIdx.x;
    if (g < nG) {
        float acc = bh2[0];
        #pragma unroll
        for (int f = 0; f < 32; ++f) acc += hid[g * 32 + f] * Wh2[f];
        y[g] = acc;
    }
}

// ------------------------------- launcher ----------------------------------

extern "C" void kernel_launch(void* const* d_in, const int* in_sizes, int n_in,
                              void* d_out, int out_size, void* d_ws, size_t ws_size,
                              hipStream_t stream) {
    const float* x     = (const float*)d_in[0];
    const int*   ei    = (const int*)  d_in[1];
    const int*   batch = (const int*)  d_in[2];
    const float* W1    = (const float*)d_in[3];
    const float* b1    = (const float*)d_in[4];
    const float* W2    = (const float*)d_in[5];
    const float* b2    = (const float*)d_in[6];
    const float* W3    = (const float*)d_in[7];
    const float* b3    = (const float*)d_in[8];
    const float* Wh1   = (const float*)d_in[9];
    const float* bh1   = (const float*)d_in[10];
    const float* Wh2   = (const float*)d_in[11];
    const float* bh2   = (const float*)d_in[12];
    float* out = (float*)d_out;

    const int N = in_sizes[0] / GCN_INDIM;   // 100000 (multiple of 16)
    const int E = in_sizes[1] / 2;           // 3200000
    const int* src = ei;                     // edge_index[0]
    const int* dst = ei + E;                 // edge_index[1]

    // workspace layout
    float*    P    = (float*)d_ws;                       // N x 64 f32 (GEMM out)
    float*    Q    = P + (size_t)N * GCN_HID;            // N x 64 f32 (agg buffer)
    _Float16* Ah   = (_Float16*)(Q + (size_t)N * GCN_HID); // N x 64 f16 acts (xh aliases first Nx32)
    float*    dinv = (float*)(Ah + (size_t)N * GCN_HID); // N
    float*    gsum = dinv + (size_t)N;                   // G x 32
    float*    gcnt = gsum + (size_t)GCN_G * 32;          // G
    float*    hid  = gcnt + (size_t)GCN_G;               // G x 32 f32
    _Float16* gh   = (_Float16*)(hid + (size_t)GCN_G * 32); // G x 32 f16

    auto cdiv = [](long a, long b) { return (int)((a + b - 1) / b); };
    const dim3 w32(32);

    // --- degrees (self-loop baked in via init=1.0) ---
    k_fill  <<<cdiv(N, 256), 256, 0, stream>>>(dinv, 1.0f, N);
    k_degree<<<cdiv(E, 256), 256, 0, stream>>>(dst, dinv, E);
    k_dinv  <<<cdiv(N, 256), 256, 0, stream>>>(dinv, N);

    // --- pad/convert x to f16 [N,32] ---
    k_xpad<<<cdiv((long)N * 32, 256), 256, 0, stream>>>(x, Ah, N);

    // --- layer 1: h = relu(agg(x @ W1) + b1) ---
    k_gemm_wmma<32, 64><<<N / 16, w32, 0, stream>>>(Ah, W1, GCN_INDIM, P, nullptr, 0);
    k_selfloop   <<<cdiv((long)N * 64, 256), 256, 0, stream>>>(P, dinv, Q, N, 6);
    k_edge_agg   <<<cdiv((long)E * 4, 256), 256, 0, stream>>>(src, dst, P, dinv, Q, E, 2, 6);
    k_bias_relu_h<<<cdiv((long)N * 64, 256), 256, 0, stream>>>(Q, Ah, b1, N, 6);

    // --- layer 2 ---
    k_gemm_wmma<64, 64><<<N / 16, w32, 0, stream>>>(Ah, W2, GCN_HID, P, nullptr, 0);
    k_selfloop   <<<cdiv((long)N * 64, 256), 256, 0, stream>>>(P, dinv, Q, N, 6);
    k_edge_agg   <<<cdiv((long)E * 4, 256), 256, 0, stream>>>(src, dst, P, dinv, Q, E, 2, 6);
    k_bias_relu_h<<<cdiv((long)N * 64, 256), 256, 0, stream>>>(Q, Ah, b2, N, 6);

    // --- layer 3 (F = 32) ---
    k_gemm_wmma<64, 32><<<N / 16, w32, 0, stream>>>(Ah, W3, GCN_HID, P, nullptr, 0);
    k_selfloop   <<<cdiv((long)N * 32, 256), 256, 0, stream>>>(P, dinv, Q, N, 5);
    k_edge_agg   <<<cdiv((long)E * 2, 256), 256, 0, stream>>>(src, dst, P, dinv, Q, E, 1, 5);
    k_bias_relu_h<<<cdiv((long)N * 32, 256), 256, 0, stream>>>(Q, Ah, b3, N, 5);

    // --- global mean pool (gsum and gcnt contiguous -> one fill) ---
    k_fill<<<cdiv(GCN_G * 32 + GCN_G, 256), 256, 0, stream>>>(gsum, 0.0f, GCN_G * 32 + GCN_G);
    k_pool_accum<<<cdiv((long)N * 2, 256), 256, 0, stream>>>(Q, batch, gsum, gcnt, N);
    k_pool_div_h<<<cdiv(GCN_G * 32, 256), 256, 0, stream>>>(gsum, gcnt, gh, GCN_G);

    // --- head: relu(g @ Wh1 + bh1) @ Wh2 + bh2 ---
    k_gemm_wmma<32, 32><<<GCN_G / 16, w32, 0, stream>>>(gh, Wh1, 32, hid, bh1, 1);
    k_head2<<<cdiv(GCN_G, 256), 256, 0, stream>>>(hid, Wh2, bh2, out, GCN_G);
}
